// IAFChainEncoder_20761871908959
// MI455X (gfx1250) — matrix operand order repaired and checked by reading, not compiled, not run on hardware
//
#include <hip/hip_runtime.h>
#include <cstdint>

// ---------------------------------------------------------------------------
// IAF chain (B=8192, D=256, H=512, T=4) for MI455X (gfx1250, wave32, WMMA).
// GEMMs run on v_wmma_f32_16x16x32_bf16 using a bf16 hi/lo split of the fp32
// operands (3 WMMAs per tile ~ fp32 accuracy at bf16-pipe rate).
// Global->LDS staging is double-buffered and, when the toolchain provides it,
// uses gfx1250 GLOBAL_LOAD_ASYNC_TO_LDS_B128 (ASYNCcnt) so the next K-tile
// streams into LDS underneath the current tile's WMMAs.
// ---------------------------------------------------------------------------

typedef __attribute__((ext_vector_type(16))) __bf16        v16bf;
typedef __attribute__((ext_vector_type(8)))  float         v8f;
typedef __attribute__((ext_vector_type(4)))  unsigned int  u4;
typedef int v4i __attribute__((vector_size(16)));          // builtin's pointee type

union FragU { v16bf v; u4 q[2]; };

constexpr int cB = 8192;
constexpr int cD = 256;
constexpr int cH = 512;
constexpr int cT = 4;
constexpr int cN3 = 2 * cD;   // = 512

#if defined(__gfx1250__) && __has_builtin(__builtin_amdgcn_global_load_async_to_lds_b128)
#define USE_ASYNC 1
#else
#define USE_ASYNC 0
#endif

#if USE_ASYNC
typedef __attribute__((address_space(1))) v4i* gv4_t;   // global v4i*
typedef __attribute__((address_space(3))) v4i* lv4_t;   // LDS v4i*

__device__ __forceinline__ void gld_async_b128(const void* g, void* l) {
    // flat LDS address low 32 bits == group-segment offset (ISA 10.2 aperture)
    __builtin_amdgcn_global_load_async_to_lds_b128(
        (gv4_t)(uintptr_t)g, (lv4_t)(uint32_t)(uintptr_t)l, 0, 0);
}
__device__ __forceinline__ void wait_async_le8() {
#if __has_builtin(__builtin_amdgcn_s_wait_asynccnt)
    __builtin_amdgcn_s_wait_asynccnt(8);
#else
    asm volatile("s_wait_asynccnt 0x8" ::: "memory");
#endif
}
__device__ __forceinline__ void wait_async_le0() {
#if __has_builtin(__builtin_amdgcn_s_wait_asynccnt)
    __builtin_amdgcn_s_wait_asynccnt(0);
#else
    asm volatile("s_wait_asynccnt 0x0" ::: "memory");
#endif
}
#endif

// ---------------------------------------------------------------------------
// x = mean + exp(0.5*log_var)*eps
__global__ void k_sample(const float* __restrict__ mean, const float* __restrict__ lv,
                         const float* __restrict__ eps, float* __restrict__ X) {
    int i = blockIdx.x * 256 + threadIdx.x;
    X[i] = mean[i] + expf(0.5f * lv[i]) * eps[i];
}

// ---------------------------------------------------------------------------
// Masked weight prep: W (T,K,N) f32 -> Wt (T,N,K) bf16 hi/lo, mask by index.
// type 1: m1[k][n] = (n%255) >= k
// type 2: m2[k][n] = (n%255) >= (k%255)
// type 3: m3[k][n] = (n>>1) >  (k%255)
__global__ void k_wprep(const float* __restrict__ W, int K, int N, int type,
                        __bf16* __restrict__ Whi, __bf16* __restrict__ Wlo) {
    int i = blockIdx.x * 256 + threadIdx.x;        // over T*K*N
    int kn = K * N;
    int t = i / kn, r = i - t * kn;
    int k = r / N, n = r - k * N;
    bool m;
    if (type == 1)      m = (n % 255) >= k;
    else if (type == 2) m = (n % 255) >= (k % 255);
    else                m = (n >> 1) > (k % 255);
    float v = m ? W[i] : 0.0f;
    __bf16 h = (__bf16)v;
    size_t o = (size_t)t * kn + (size_t)n * K + k; // transposed: N-major
    Whi[o] = h;
    Wlo[o] = (__bf16)(v - (float)h);
}

// ---------------------------------------------------------------------------
// Reverse + bf16 hi/lo split of activations: XR[b][d] = X[b][D-1-d]
__global__ void k_actprep(const float* __restrict__ X,
                          __bf16* __restrict__ XRhi, __bf16* __restrict__ XRlo) {
    int i = blockIdx.x * 256 + threadIdx.x;        // over B*D
    int b = i >> 8, d = i & 255;
    float v = X[(b << 8) + (255 - d)];
    __bf16 h = (__bf16)v;
    XRhi[i] = h;
    XRlo[i] = (__bf16)(v - (float)h);
}

// ---------------------------------------------------------------------------
// x_new[b][d] = (xr - shift) * exp(-tanh(log_var_raw))
__global__ void k_update(const float* __restrict__ X, const float* __restrict__ OUT,
                         float* __restrict__ Xn) {
    int i = blockIdx.x * 256 + threadIdx.x;        // over B*D
    int b = i >> 8, d = i & 255;
    float xr = X[(b << 8) + (255 - d)];
    float s  = OUT[(b << 9) + 2 * d];
    float ls = tanhf(OUT[(b << 9) + 2 * d + 1]);
    Xn[i] = (xr - s) * expf(-ls);
}

// ---------------------------------------------------------------------------
// Stage one 128x32 K-tile of all four operand arrays into one LDS buffer.
// 512 16B chunks per array / 256 threads = 2 chunks each; 8 transfers/thread.
__device__ __forceinline__ void stage_tile(
    const __bf16* __restrict__ Ahi, const __bf16* __restrict__ Alo,
    const __bf16* __restrict__ Bthi, const __bf16* __restrict__ Btlo,
    int K, int rowBase, int colBase, int k0, int tid,
    __bf16* dAh, __bf16* dAl, __bf16* dBh, __bf16* dBl) {
#pragma unroll
    for (int p = 0; p < 2; ++p) {
        int c  = tid + p * 256;
        int r  = c >> 2;
        int kc = (c & 3) * 8;
        size_t ga = (size_t)(rowBase + r) * K + k0 + kc;
        size_t gb = (size_t)(colBase + r) * K + k0 + kc;
        int lo = r * 40 + kc;
#if USE_ASYNC
        gld_async_b128(&Ahi[ga],  &dAh[lo]);
        gld_async_b128(&Alo[ga],  &dAl[lo]);
        gld_async_b128(&Bthi[gb], &dBh[lo]);
        gld_async_b128(&Btlo[gb], &dBl[lo]);
#else
        *(u4*)&dAh[lo] = *(const u4*)&Ahi[ga];
        *(u4*)&dAl[lo] = *(const u4*)&Alo[ga];
        *(u4*)&dBh[lo] = *(const u4*)&Bthi[gb];
        *(u4*)&dBl[lo] = *(const u4*)&Btlo[gb];
#endif
    }
}

// ---------------------------------------------------------------------------
// GEMM: C[M,N] = A[M,K] @ B[K,N] (+bias), A/B given as bf16 hi/lo pairs,
// B pre-transposed to [N,K]. Block tile 128x128, BK=32, 8 waves of 64x32,
// double-buffered LDS with async prefetch of the next K-tile.
// EPI==0: relu + bf16 hi/lo resplit stores; EPI==1: fp32 store.
template <int EPI>
__global__ __launch_bounds__(256)
void k_gemm(const __bf16* __restrict__ Ahi, const __bf16* __restrict__ Alo,
            const __bf16* __restrict__ Bthi, const __bf16* __restrict__ Btlo,
            const float* __restrict__ bias, int K, int N,
            __bf16* __restrict__ Chi, __bf16* __restrict__ Clo,
            float* __restrict__ Cf) {
    // +8 bf16 (16B) row pad -> conflict-free ds_read_b128 fragment reads
    __shared__ __align__(16) __bf16 sAh[2][128 * 40];
    __shared__ __align__(16) __bf16 sAl[2][128 * 40];
    __shared__ __align__(16) __bf16 sBh[2][128 * 40];
    __shared__ __align__(16) __bf16 sBl[2][128 * 40];

    const int tid  = threadIdx.x;
    const int lane = tid & 31;
    const int w    = tid >> 5;
    const int half = lane >> 4;        // 0: lanes 0-15, 1: lanes 16-31
    const int lr   = lane & 15;
    const int rowBase = blockIdx.x * 128;
    const int colBase = blockIdx.y * 128;
    const int wmoff = (w & 1) * 64;    // wave M offset in block
    const int wnoff = (w >> 1) * 32;   // wave N offset in block

    v8f acc[4][2] = {};

    const int nk = K >> 5;
    // prologue: fill buffer 0
    stage_tile(Ahi, Alo, Bthi, Btlo, K, rowBase, colBase, 0, tid,
               sAh[0], sAl[0], sBh[0], sBl[0]);

    for (int ki = 0; ki < nk; ++ki) {
        const int cur = ki & 1;
        const int nxt = cur ^ 1;
        if (ki + 1 < nk) {
            // prefetch next K-tile into the other buffer (overlaps compute)
            stage_tile(Ahi, Alo, Bthi, Btlo, K, rowBase, colBase, (ki + 1) << 5, tid,
                       sAh[nxt], sAl[nxt], sBh[nxt], sBl[nxt]);
#if USE_ASYNC
            wait_async_le8();   // first 8 (cur tile) done; 8 newest still in flight
        } else {
            wait_async_le0();
#endif
        }
        __syncthreads();        // cur buffer visible to all waves

        // A fragments (16x32 bf16): lanes<16 K{0..7,16..23}, lanes>=16 K{8..15,24..31}
        FragU ah[4], al[4], bh[2], bl[2];
#pragma unroll
        for (int mt = 0; mt < 4; ++mt) {
            int r = wmoff + mt * 16 + lr;
            ah[mt].q[0] = *(const u4*)&sAh[cur][r * 40 + half * 8];
            ah[mt].q[1] = *(const u4*)&sAh[cur][r * 40 + half * 8 + 16];
            al[mt].q[0] = *(const u4*)&sAl[cur][r * 40 + half * 8];
            al[mt].q[1] = *(const u4*)&sAl[cur][r * 40 + half * 8 + 16];
        }
        // B fragments (32x16 bf16): per-lane column, lanes<16 K{0..7,16..23}, lanes>=16 K{8..15,24..31}
#pragma unroll
        for (int nt = 0; nt < 2; ++nt) {
            int cn = wnoff + nt * 16 + lr;
            bh[nt].q[0] = *(const u4*)&sBh[cur][cn * 40 + half * 8];
            bh[nt].q[1] = *(const u4*)&sBh[cur][cn * 40 + half * 8 + 16];
            bl[nt].q[0] = *(const u4*)&sBl[cur][cn * 40 + half * 8];
            bl[nt].q[1] = *(const u4*)&sBl[cur][cn * 40 + half * 8 + 16];
        }
        // bf16x3: hi*hi + hi*lo + lo*hi  (~fp32 accuracy)
#pragma unroll
        for (int mt = 0; mt < 4; ++mt)
#pragma unroll
            for (int nt = 0; nt < 2; ++nt) {
                acc[mt][nt] = __builtin_amdgcn_wmma_f32_16x16x32_bf16(
                    false, ah[mt].v, false, bh[nt].v, (short)0, acc[mt][nt], false, false);
                acc[mt][nt] = __builtin_amdgcn_wmma_f32_16x16x32_bf16(
                    false, ah[mt].v, false, bl[nt].v, (short)0, acc[mt][nt], false, false);
                acc[mt][nt] = __builtin_amdgcn_wmma_f32_16x16x32_bf16(
                    false, al[mt].v, false, bh[nt].v, (short)0, acc[mt][nt], false, false);
            }
        __syncthreads();        // all waves done with cur before it is refilled
    }

    // Epilogue: C layout -> VGPR j, lanes<16: M=j, lanes>=16: M=j+8; N=lane%16
#pragma unroll
    for (int nt = 0; nt < 2; ++nt) {
        int col = colBase + wnoff + nt * 16 + lr;
        float bv = bias[col];
#pragma unroll
        for (int mt = 0; mt < 4; ++mt) {
#pragma unroll
            for (int j = 0; j < 8; ++j) {
                int row = rowBase + wmoff + mt * 16 + j + half * 8;
                float v = acc[mt][nt][j] + bv;
                size_t o = (size_t)row * N + col;
                if constexpr (EPI == 0) {
                    v = fmaxf(v, 0.0f);
                    __bf16 h = (__bf16)v;
                    Chi[o] = h;
                    Clo[o] = (__bf16)(v - (float)h);
                } else {
                    Cf[o] = v;
                }
            }
        }
    }
}

// ---------------------------------------------------------------------------
extern "C" void kernel_launch(void* const* d_in, const int* in_sizes, int n_in,
                              void* d_out, int out_size, void* d_ws, size_t ws_size,
                              hipStream_t stream) {
    const float* mean = (const float*)d_in[0];
    const float* logv = (const float*)d_in[1];
    const float* eps  = (const float*)d_in[2];
    const float* W1   = (const float*)d_in[3];
    const float* b1   = (const float*)d_in[4];
    const float* W2   = (const float*)d_in[5];
    const float* b2   = (const float*)d_in[6];
    const float* W3   = (const float*)d_in[7];
    const float* b3   = (const float*)d_in[8];
    (void)in_sizes; (void)n_in; (void)out_size; (void)ws_size;

    char* ws = (char*)d_ws;
    size_t off = 0;
    auto alloc = [&](size_t bytes) -> void* {
        off = (off + 255) & ~(size_t)255;
        void* p = ws + off;
        off += bytes;
        return p;
    };

    float*  X0   = (float*) alloc((size_t)cB * cD * 4);
    float*  X1   = (float*) alloc((size_t)cB * cD * 4);
    __bf16* XRhi = (__bf16*)alloc((size_t)cB * cD * 2);
    __bf16* XRlo = (__bf16*)alloc((size_t)cB * cD * 2);
    __bf16* H1h  = (__bf16*)alloc((size_t)cB * cH * 2);
    __bf16* H1l  = (__bf16*)alloc((size_t)cB * cH * 2);
    __bf16* H2h  = (__bf16*)alloc((size_t)cB * cH * 2);
    __bf16* H2l  = (__bf16*)alloc((size_t)cB * cH * 2);
    float*  OUT  = (float*) alloc((size_t)cB * cN3 * 4);
    __bf16* W1h  = (__bf16*)alloc((size_t)cT * cD * cH * 2);
    __bf16* W1l  = (__bf16*)alloc((size_t)cT * cD * cH * 2);
    __bf16* W2h  = (__bf16*)alloc((size_t)cT * cH * cH * 2);
    __bf16* W2l  = (__bf16*)alloc((size_t)cT * cH * cH * 2);
    __bf16* W3h  = (__bf16*)alloc((size_t)cT * cH * cN3 * 2);
    __bf16* W3l  = (__bf16*)alloc((size_t)cT * cH * cN3 * 2);

    const int thr = 256;
    k_sample<<<(cB * cD) / thr, thr, 0, stream>>>(mean, logv, eps, X0);
    k_wprep<<<(cT * cD * cH) / thr, thr, 0, stream>>>(W1, cD, cH, 1, W1h, W1l);
    k_wprep<<<(cT * cH * cH) / thr, thr, 0, stream>>>(W2, cH, cH, 2, W2h, W2l);
    k_wprep<<<(cT * cH * cN3) / thr, thr, 0, stream>>>(W3, cH, cN3, 3, W3h, W3l);

    dim3 grid(cB / 128, cH / 128);   // N=512 for all three layers
    float* Xc = X0;
    float* Xn = X1;
    for (int it = cT - 1; it >= 0; --it) {
        k_actprep<<<(cB * cD) / thr, thr, 0, stream>>>(Xc, XRhi, XRlo);
        k_gemm<0><<<grid, thr, 0, stream>>>(
            XRhi, XRlo, W1h + (size_t)it * cD * cH, W1l + (size_t)it * cD * cH,
            b1 + it * cH, cD, cH, H1h, H1l, nullptr);
        k_gemm<0><<<grid, thr, 0, stream>>>(
            H1h, H1l, W2h + (size_t)it * cH * cH, W2l + (size_t)it * cH * cH,
            b2 + it * cH, cH, cH, H2h, H2l, nullptr);
        k_gemm<1><<<grid, thr, 0, stream>>>(
            H2h, H2l, W3h + (size_t)it * cH * cN3, W3l + (size_t)it * cH * cN3,
            b3 + it * cN3, cH, cN3, nullptr, nullptr, OUT);
        float* dst = (it == 0) ? (float*)d_out : Xn;
        k_update<<<(cB * cD) / thr, thr, 0, stream>>>(Xc, OUT, dst);
        float* t = Xc; Xc = Xn; Xn = t;
    }
}
